// Attention2d_7507602833505
// MI455X (gfx1250) — matrix operand order repaired and checked by reading, not compiled, not run on hardware
//
#include <hip/hip_runtime.h>
#include <hip/hip_bf16.h>

// ---------------------------------------------------------------------------
// Attention2d for MI455X (gfx1250): bf16 WMMA pipeline, fp32 accumulate.
// Pre-transposed operand layouts so ALL hot-loop LDS staging is contiguous
// 16B chunks -> global_load_async_to_lds_b128 (ASYNCcnt) where available.
// ---------------------------------------------------------------------------

typedef __bf16       bf16x16 __attribute__((ext_vector_type(16)));
typedef float        f32x8   __attribute__((ext_vector_type(8)));
typedef unsigned int u32x4   __attribute__((ext_vector_type(4)));
typedef int          v4i_vs  __attribute__((__vector_size__(16)));  // builtin pointee type

union FragU { u32x4 q[2]; bf16x16 v; };

#define LDS_PITCH 40   // elements; 80B row pitch: 16B aligned, conflict-free frag loads

#if defined(__HIP_DEVICE_COMPILE__) && \
    __has_builtin(__builtin_amdgcn_global_load_async_to_lds_b128)
#define ASYNC_LDS_COPY 1
#endif

// 16-byte global -> LDS copy; async (ASYNCcnt) on gfx1250 if builtin exists.
__device__ __forceinline__ void cp16(const __bf16* g, __bf16* l) {
#ifdef ASYNC_LDS_COPY
    __builtin_amdgcn_global_load_async_to_lds_b128(
        (__attribute__((address_space(1))) v4i_vs*)(unsigned long long)(const void*)g,
        (__attribute__((address_space(3))) v4i_vs*)(unsigned)(unsigned long long)(void*)l,
        0, 0);
#else
    *(u32x4*)l = *(const u32x4*)g;
#endif
}

__device__ __forceinline__ void cp16_wait() {
#ifdef ASYNC_LDS_COPY
#if __has_builtin(__builtin_amdgcn_s_wait_asynccnt)
    __builtin_amdgcn_s_wait_asynccnt(0);
#else
    asm volatile("s_wait_asynccnt 0x0" ::: "memory");
#endif
#endif
}

// A-matrix fragment (16x32 bf16): lane half h reads K[8h..8h+8) and K[16+8h..+8)
__device__ __forceinline__ bf16x16 load_fragA(const __bf16* row, int half) {
    FragU f;
    f.q[0] = *(const u32x4*)(row + half * 8);
    f.q[1] = *(const u32x4*)(row + 16 + half * 8);
    return f.v;
}
// B-matrix fragment (32x16 bf16, staged transposed [n][k]): lane half h reads K[8h..8h+8) and K[16+8h..+8)
__device__ __forceinline__ bf16x16 load_fragB(const __bf16* row, int half) {
    FragU f;
    f.q[0] = *(const u32x4*)(row + half * 8);
    f.q[1] = *(const u32x4*)(row + 16 + half * 8);
    return f.v;
}

__device__ __forceinline__ f32x8 wmma_bf16(bf16x16 a, bf16x16 b, f32x8 c) {
    return __builtin_amdgcn_wmma_f32_16x16x32_bf16(false, a, false, b, (short)0, c,
                                                   false, false);
}

// ---------------------------------------------------------------------------
// Fused f32 -> bf16 convert + 32x32 tiled transpose: in (R,C) -> out (C,R).
__global__ __launch_bounds__(256) void cvt_transpose_kernel(
    const float* __restrict__ in, __bf16* __restrict__ out,
    int R, int C, long inBS, long outBS) {
    __shared__ __bf16 tile[32][40];
    const float* ib = in + inBS * blockIdx.z;
    __bf16* ob = out + outBS * blockIdx.z;
    const int c0 = blockIdx.x * 32, r0 = blockIdx.y * 32;
    const int tx = threadIdx.x & 31, ty = threadIdx.x >> 5;  // 32 x 8
#pragma unroll
    for (int i = 0; i < 32; i += 8)
        tile[ty + i][tx] = (__bf16)ib[(long)(r0 + ty + i) * C + c0 + tx];
    __syncthreads();
#pragma unroll
    for (int i = 0; i < 32; i += 8)
        ob[(long)(c0 + ty + i) * R + r0 + tx] = tile[tx][ty + i];
}

__global__ __launch_bounds__(256) void zero_u32_kernel(unsigned int* __restrict__ p, int n) {
    int i = blockIdx.x * 256 + threadIdx.x;
    if (i < n) p[i] = 0u;
}

// ---------------------------------------------------------------------------
// QKV GEMM:  qkv[b,d,s] = sum_c wqT[d,c] * xT[b,s,c] + b_qkv[d]
// M=2304 (d), N=1024 (s), K=768 (c). Epilogue scatters into attention
// layouts: Qs/Ks (b,h,1024,32) seq-major, Vd (b,h,32,1024) d-major
// (pad cols/rows 24..31 pre-zeroed).
__global__ __launch_bounds__(256) void qkv_gemm_kernel(
    const __bf16* __restrict__ xT, const __bf16* __restrict__ wqT,
    const float* __restrict__ bqkv, __bf16* __restrict__ Qs,
    __bf16* __restrict__ Ks, __bf16* __restrict__ Vd) {
    __shared__ __bf16 At[64 * LDS_PITCH];    // [m][k]
    __shared__ __bf16 Bt[128 * LDS_PITCH];   // [n][k]

    const int t = threadIdx.x;
    const int lane = t & 31, wv = t >> 5;
    const int half = lane >> 4, l16 = lane & 15;
    const int Mb = blockIdx.x * 64, Nb = blockIdx.y * 128, b = blockIdx.z;
    const int wm = wv & 1, wn = wv >> 1;            // 2x4 wave grid, 32x32 per wave
    const __bf16* xrow = xT + (long)b * 1024 * 768;

    f32x8 acc[2][2];
    f32x8 zero = {0.f, 0.f, 0.f, 0.f, 0.f, 0.f, 0.f, 0.f};
    acc[0][0] = zero; acc[0][1] = zero; acc[1][0] = zero; acc[1][1] = zero;

    // staging index precompute (contiguous 16B chunks)
    const int am = t >> 2, ac = t & 3;              // A: 64 rows x 4 chunks

    for (int k0 = 0; k0 < 768; k0 += 32) {
        cp16(wqT + (long)(Mb + am) * 768 + k0 + ac * 8, At + am * LDS_PITCH + ac * 8);
#pragma unroll
        for (int s2 = 0; s2 < 2; s2++) {            // B: 128 rows x 4 chunks
            int idx = t + s2 * 256;
            int bn = idx >> 2, bc = idx & 3;
            cp16(xrow + (long)(Nb + bn) * 768 + k0 + bc * 8, Bt + bn * LDS_PITCH + bc * 8);
        }
        cp16_wait();
        __syncthreads();

        bf16x16 Af[2], Bf[2];
#pragma unroll
        for (int mt = 0; mt < 2; mt++)
            Af[mt] = load_fragA(At + (wm * 32 + mt * 16 + l16) * LDS_PITCH, half);
#pragma unroll
        for (int nt = 0; nt < 2; nt++)
            Bf[nt] = load_fragB(Bt + (wn * 32 + nt * 16 + l16) * LDS_PITCH, half);
#pragma unroll
        for (int mt = 0; mt < 2; mt++)
#pragma unroll
            for (int nt = 0; nt < 2; nt++)
                acc[mt][nt] = wmma_bf16(Af[mt], Bf[nt], acc[mt][nt]);
        __syncthreads();
    }

    // epilogue: bias + scatter
#pragma unroll
    for (int mt = 0; mt < 2; mt++) {
#pragma unroll
        for (int r = 0; r < 8; r++) {
            int d = Mb + wm * 32 + mt * 16 + r + 8 * half;
            float bias = bqkv[d];
            int p = d / 768, rem = d % 768;
            int head = rem / 24, rr = rem % 24;
            long bh = (long)b * 32 + head;
#pragma unroll
            for (int nt = 0; nt < 2; nt++) {
                int s = Nb + wn * 32 + nt * 16 + l16;
                __bf16 v = (__bf16)(acc[mt][nt][r] + bias);
                if (p == 0)      Qs[(bh * 1024 + s) * 32 + rr] = v;
                else if (p == 1) Ks[(bh * 1024 + s) * 32 + rr] = v;
                else             Vd[(bh * 32 + rr) * 1024 + s] = v;
            }
        }
    }
}

// ---------------------------------------------------------------------------
// Flash attention per (b, head, 128-query tile). hd padded 24 -> 32.
// Each of 8 waves owns 16 queries; streams 32-key blocks.
__global__ __launch_bounds__(256) void attn_kernel(
    const __bf16* __restrict__ Qs, const __bf16* __restrict__ Ks,
    const __bf16* __restrict__ Vd, __bf16* __restrict__ oT) {
    __shared__ __bf16 Qt[128 * LDS_PITCH];       // [q][d]
    __shared__ __bf16 Kt[32 * LDS_PITCH];        // [k][d]
    __shared__ __bf16 Vt[32 * LDS_PITCH];        // [d][k]
    __shared__ __bf16 Ps[8 * 16 * LDS_PITCH];    // per-wave P scratch [m][k]

    const int t = threadIdx.x;
    const int lane = t & 31, wv = t >> 5;
    const int half = lane >> 4, l16 = lane & 15;
    const int q0 = blockIdx.x * 128;
    const int bh = blockIdx.y;                    // b*32 + head
    const int b = bh >> 5, head = bh & 31;

    const __bf16* Qg = Qs + (long)bh * 1024 * 32;
    const __bf16* Kg = Ks + (long)bh * 1024 * 32;
    const __bf16* Vg = Vd + (long)bh * 32 * 1024;

    // stage Q [q][d]: 128 rows x 4 chunks = 512, 2 per thread
#pragma unroll
    for (int s2 = 0; s2 < 2; s2++) {
        int idx = t + s2 * 256;
        int q = idx >> 2, c = idx & 3;
        cp16(Qg + (long)(q0 + q) * 32 + c * 8, Qt + q * LDS_PITCH + c * 8);
    }
    cp16_wait();
    __syncthreads();

    bf16x16 Aq = load_fragA(Qt + (wv * 16 + l16) * LDS_PITCH, half);

    f32x8 O0 = {0.f, 0.f, 0.f, 0.f, 0.f, 0.f, 0.f, 0.f};
    f32x8 O1 = O0;
    float mrun[8], lrun[8];
#pragma unroll
    for (int r = 0; r < 8; r++) { mrun[r] = -3.0e38f; lrun[r] = 0.f; }
    const float scale = 0.20412414523193154f;  // 24^-0.5

    __bf16* myP = Ps + wv * 16 * LDS_PITCH;

    for (int k0 = 0; k0 < 1024; k0 += 32) {
        if (t < 128) {                      // stage K [k][d]: 32 rows x 4 chunks
            int kk = t >> 2, c = t & 3;
            cp16(Kg + (long)(k0 + kk) * 32 + c * 8, Kt + kk * LDS_PITCH + c * 8);
        } else {                            // stage V [d][k]: 32 rows x 4 chunks
            int u = t - 128;
            int dd = u >> 2, c = u & 3;
            cp16(Vg + (long)dd * 1024 + k0 + c * 8, Vt + dd * LDS_PITCH + c * 8);
        }
        cp16_wait();
        __syncthreads();

        // S = Q^T K  (16 q x 32 k), two N-tiles
        bf16x16 Bk0 = load_fragB(Kt + l16 * LDS_PITCH, half);
        bf16x16 Bk1 = load_fragB(Kt + (16 + l16) * LDS_PITCH, half);
        f32x8 zero = {0.f, 0.f, 0.f, 0.f, 0.f, 0.f, 0.f, 0.f};
        f32x8 S0 = wmma_bf16(Aq, Bk0, zero);
        f32x8 S1 = wmma_bf16(Aq, Bk1, zero);

        float als[8];
#pragma unroll
        for (int r = 0; r < 8; r++) {
            float s0 = S0[r] * scale, s1 = S1[r] * scale;
            float rm = fmaxf(s0, s1);
#pragma unroll
            for (int off = 8; off >= 1; off >>= 1)
                rm = fmaxf(rm, __shfl_xor(rm, off, 16));
            float mn = fmaxf(mrun[r], rm);
            float al = __expf(mrun[r] - mn);
            float p0 = __expf(s0 - mn), p1 = __expf(s1 - mn);
            float rs = p0 + p1;
#pragma unroll
            for (int off = 8; off >= 1; off >>= 1)
                rs += __shfl_xor(rs, off, 16);
            lrun[r] = lrun[r] * al + rs;
            mrun[r] = mn;
            als[r] = al;
            int m = r + 8 * half;
            myP[m * LDS_PITCH + l16] = (__bf16)p0;
            myP[m * LDS_PITCH + 16 + l16] = (__bf16)p1;
        }
#pragma unroll
        for (int r = 0; r < 8; r++) { O0[r] *= als[r]; O1[r] *= als[r]; }

        // O += P * V^T   (A=P 16x32, B=V [d][k] two N-tiles)
        bf16x16 Ap = load_fragA(myP + l16 * LDS_PITCH, half);
        bf16x16 Bv0 = load_fragB(Vt + l16 * LDS_PITCH, half);
        bf16x16 Bv1 = load_fragB(Vt + (16 + l16) * LDS_PITCH, half);
        O0 = wmma_bf16(Ap, Bv0, O0);
        O1 = wmma_bf16(Ap, Bv1, O1);
        __syncthreads();
    }

    // epilogue: normalize, drop pad rows, write oT[b, q, head*24+d] bf16 (seq-major)
#pragma unroll
    for (int r = 0; r < 8; r++) {
        float inv = 1.0f / lrun[r];
        int q = q0 + wv * 16 + r + 8 * half;
        long rowoff = ((long)b * 1024 + q) * 768 + head * 24;
        int d0 = l16;
        if (d0 < 24) oT[rowoff + d0] = (__bf16)(O0[r] * inv);
        int d1 = l16 + 16;
        if (d1 < 24) oT[rowoff + d1] = (__bf16)(O1[r] * inv);
    }
}

// ---------------------------------------------------------------------------
// Proj GEMM: out[b,f,s] = sum_c wpT[f,c] * oT[b,s,c] + b_proj[f]  (f32 out)
__global__ __launch_bounds__(256) void proj_gemm_kernel(
    const __bf16* __restrict__ oT, const __bf16* __restrict__ wpT,
    const float* __restrict__ bproj, float* __restrict__ out) {
    __shared__ __bf16 At[64 * LDS_PITCH];
    __shared__ __bf16 Bt[128 * LDS_PITCH];

    const int t = threadIdx.x;
    const int lane = t & 31, wv = t >> 5;
    const int half = lane >> 4, l16 = lane & 15;
    const int Mb = blockIdx.x * 64, Nb = blockIdx.y * 128, b = blockIdx.z;
    const int wm = wv & 1, wn = wv >> 1;
    const __bf16* orow = oT + (long)b * 1024 * 768;

    f32x8 acc[2][2];
    f32x8 zero = {0.f, 0.f, 0.f, 0.f, 0.f, 0.f, 0.f, 0.f};
    acc[0][0] = zero; acc[0][1] = zero; acc[1][0] = zero; acc[1][1] = zero;

    const int am = t >> 2, ac = t & 3;

    for (int k0 = 0; k0 < 768; k0 += 32) {
        cp16(wpT + (long)(Mb + am) * 768 + k0 + ac * 8, At + am * LDS_PITCH + ac * 8);
#pragma unroll
        for (int s2 = 0; s2 < 2; s2++) {
            int idx = t + s2 * 256;
            int bn = idx >> 2, bc = idx & 3;
            cp16(orow + (long)(Nb + bn) * 768 + k0 + bc * 8, Bt + bn * LDS_PITCH + bc * 8);
        }
        cp16_wait();
        __syncthreads();

        bf16x16 Af[2], Bf[2];
#pragma unroll
        for (int mt = 0; mt < 2; mt++)
            Af[mt] = load_fragA(At + (wm * 32 + mt * 16 + l16) * LDS_PITCH, half);
#pragma unroll
        for (int nt = 0; nt < 2; nt++)
            Bf[nt] = load_fragB(Bt + (wn * 32 + nt * 16 + l16) * LDS_PITCH, half);
#pragma unroll
        for (int mt = 0; mt < 2; mt++)
#pragma unroll
            for (int nt = 0; nt < 2; nt++)
                acc[mt][nt] = wmma_bf16(Af[mt], Bf[nt], acc[mt][nt]);
        __syncthreads();
    }

#pragma unroll
    for (int mt = 0; mt < 2; mt++) {
#pragma unroll
        for (int r = 0; r < 8; r++) {
            int f = Mb + wm * 32 + mt * 16 + r + 8 * half;
            float bias = bproj[f];
#pragma unroll
            for (int nt = 0; nt < 2; nt++) {
                int s = Nb + wn * 32 + nt * 16 + l16;
                out[((long)b * 768 + f) * 1024 + s] = acc[mt][nt][r] + bias;
            }
        }
    }
}

// ---------------------------------------------------------------------------
extern "C" void kernel_launch(void* const* d_in, const int* in_sizes, int n_in,
                              void* d_out, int out_size, void* d_ws, size_t ws_size,
                              hipStream_t stream) {
    const float* x      = (const float*)d_in[0];  // (4,768,32,32)
    const float* w_qkv  = (const float*)d_in[1];  // (768,2304)
    const float* b_qkv  = (const float*)d_in[2];  // (2304,)
    const float* w_proj = (const float*)d_in[3];  // (768,768)
    const float* b_proj = (const float*)d_in[4];  // (768,)
    float* out = (float*)d_out;                   // (4,768,32,32) f32

    char* w = (char*)d_ws;                        // 256B-aligned segments
    __bf16* xT  = (__bf16*)(w);                   //  6,291,456 B  (b,s,c)
    __bf16* wqT = (__bf16*)(w + 6291456);         //  3,538,944 B  (d,c)
    __bf16* wpT = (__bf16*)(w + 9830400);         //  1,179,648 B  (f,c)
    __bf16* Qs  = (__bf16*)(w + 11010048);        //  8,388,608 B  (b,h,s,32)
    __bf16* Ks  = (__bf16*)(w + 19398656);        //  8,388,608 B  (b,h,s,32)
    __bf16* Vd  = (__bf16*)(w + 27787264);        //  8,388,608 B  (b,h,32,s)
    __bf16* oT  = (__bf16*)(w + 36175872);        //  6,291,456 B  (b,s,c)  -> 42.5 MB

    // transpose+convert pre-pass
    cvt_transpose_kernel<<<dim3(32, 24, 4), 256, 0, stream>>>(
        x, xT, 768, 1024, 768L * 1024, 1024L * 768);
    cvt_transpose_kernel<<<dim3(72, 24, 1), 256, 0, stream>>>(
        w_qkv, wqT, 768, 2304, 0, 0);
    cvt_transpose_kernel<<<dim3(24, 24, 1), 256, 0, stream>>>(
        w_proj, wpT, 768, 768, 0, 0);

    // zero Qs/Ks/Vd (contiguous) so pad lanes 24..31 contribute nothing
    const int nz = (3 * 8388608) / 4;
    zero_u32_kernel<<<(nz + 255) / 256, 256, 0, stream>>>((unsigned int*)Qs, nz);

    qkv_gemm_kernel<<<dim3(36, 8, 4), 256, 0, stream>>>(xT, wqT, b_qkv, Qs, Ks, Vd);
    attn_kernel<<<dim3(8, 128), 256, 0, stream>>>(Qs, Ks, Vd, oT);
    proj_gemm_kernel<<<dim3(12, 8, 4), 256, 0, stream>>>(oT, wpT, b_proj, out);
}
